// _RPNFasterRCNN_69887707841002
// MI455X (gfx1250) — matrix operand order repaired and checked by reading, not compiled, not run on hardware
//
#include <hip/hip_runtime.h>
#include <cstdint>

typedef __attribute__((ext_vector_type(16))) _Float16 v16h;
typedef __attribute__((ext_vector_type(8)))  _Float16 v8h;
typedef __attribute__((ext_vector_type(8)))  float    v8f;

namespace {
constexpr int kFH = 128, kFW = 128;
constexpr int kNPix = kFH * kFW;            // 16384
constexpr int kCin = 512, kCmid = 512;
constexpr int kK9  = kCin * 9;              // 4608
constexpr int kNAnch = kNPix * 9;           // 147456
constexpr int kPre = 12000, kPost = 2000;
constexpr int kMHead = 64;                  // 36 loc + 18 score, padded to 64

// ---- workspace layout (bytes, all offsets 256B aligned) ----
constexpr size_t SZ_IN_T = (size_t)kNPix * kCin * 2;     // f16 [pix][ci]
constexpr size_t SZ_W_T  = (size_t)kCmid * kK9 * 2;      // f16 [co][r*512+ci]
constexpr size_t SZ_WC   = (size_t)kMHead * kCmid * 2;   // f16 [m][ci]
constexpr size_t SZ_BC   = 256;                          // f32 [64]
constexpr size_t SZ_H_T  = (size_t)kNPix * kCmid * 2;    // f16 [pix][co]
constexpr size_t SZ_LOC  = (size_t)kNPix * 36 * 4;       // f32
constexpr size_t SZ_SCR  = (size_t)kNPix * 18 * 4;       // f32
constexpr size_t SZ_BOX  = (size_t)kNAnch * 16;          // f32 x4
constexpr size_t SZ_SC   = (size_t)kNAnch * 4;           // f32
constexpr size_t SZ_CBOX = 192256;                       // 12000 x float4 (+pad)
constexpr size_t SZ_CSC  = 48128;                        // 12000 floats (+pad)
constexpr size_t SZ_HIST = 4096;                         // 1024 ints

constexpr size_t OFF_IN_T = 0;
constexpr size_t OFF_W_T  = OFF_IN_T + SZ_IN_T;
constexpr size_t OFF_WC   = OFF_W_T + SZ_W_T;
constexpr size_t OFF_BC   = OFF_WC + SZ_WC;
constexpr size_t OFF_H_T  = OFF_BC + SZ_BC;
constexpr size_t OFF_LOC  = OFF_H_T + SZ_H_T;
constexpr size_t OFF_SCR  = OFF_LOC + SZ_LOC;
constexpr size_t OFF_BOX  = OFF_SCR + SZ_SCR;
constexpr size_t OFF_SC   = OFF_BOX + SZ_BOX;
constexpr size_t OFF_CBOX = OFF_SC + SZ_SC;
constexpr size_t OFF_CSC  = OFF_CBOX + SZ_CBOX;
constexpr size_t OFF_HIST = OFF_CSC + SZ_CSC;
constexpr size_t OFF_MISC = OFF_HIST + SZ_HIST;
// misc ints: [0]=cand count, [1]=thr bin, f[2]=sl1 sum, f[3]=nll sum, f[4]=n_valid
// bytes [32..48): 16B zero pad (async-load source for halo lanes)
} // namespace

// ---- CDNA5 async global->LDS copy (ASYNCcnt-tracked, no VGPR round trip) ----
__device__ __forceinline__ void async_load_b128(unsigned lds_addr, const void* gaddr) {
  asm volatile("global_load_async_to_lds_b128 %0, %1, off"
               :: "v"(lds_addr), "v"((unsigned long long)(uintptr_t)gaddr)
               : "memory");
}
#if __has_builtin(__builtin_amdgcn_s_wait_asynccnt)
#define WAIT_ASYNC(n) __builtin_amdgcn_s_wait_asynccnt(n)
#else
#define WAIT_ASYNC(n) asm volatile("s_wait_asynccnt %0" :: "n"(n) : "memory")
#endif

__device__ __forceinline__ unsigned lds_lo32(const void* p) {
  // flat shared-aperture address: low 32 bits == workgroup-relative LDS offset
  return (unsigned)(uintptr_t)p;
}

// ---------------------------------------------------------------------------
// Kernel 0: convert/transposes + zero-init of accumulators (grid-stride)
// ---------------------------------------------------------------------------
__global__ void rpn_prep(const float* __restrict__ feat, const float* __restrict__ convw,
                         const float* __restrict__ locw, const float* __restrict__ scrw,
                         const float* __restrict__ locb, const float* __restrict__ scrb,
                         _Float16* __restrict__ in_t, _Float16* __restrict__ w_t,
                         _Float16* __restrict__ wc, float* __restrict__ bc,
                         float* __restrict__ cbox, float* __restrict__ csc,
                         int* __restrict__ hist, int* __restrict__ misc) {
  const long long R0 = (long long)kNPix * kCin;   // in_t transpose
  const long long R1 = (long long)kCmid * kK9;    // weight transpose
  const long long R2 = (long long)kMHead * kCmid; // head weights
  const long long R3 = kMHead;                    // head bias
  const long long R4 = kPre;                      // candidate init
  const long long R5 = 1024;                      // hist
  const long long R6 = 16;                        // misc + zero pad
  const long long total = R0 + R1 + R2 + R3 + R4 + R5 + R6;
  for (long long i = (long long)blockIdx.x * 256 + threadIdx.x; i < total;
       i += (long long)gridDim.x * 256) {
    if (i < R0) {
      long long p = i >> 9; int ci = (int)(i & 511);
      in_t[i] = (_Float16)feat[(long long)ci * kNPix + p];
    } else if (i < R0 + R1) {
      long long j = i - R0;
      int co = (int)(j / kK9); int rem = (int)(j - (long long)co * kK9);
      int r = rem >> 9; int ci = rem & 511;
      w_t[j] = (_Float16)convw[((long long)co * 512 + ci) * 9 + r];
    } else if (i < R0 + R1 + R2) {
      long long j = i - R0 - R1;
      int m = (int)(j >> 9); int k = (int)(j & 511);
      float v = 0.f;
      if (m < 36)      v = locw[m * 512 + k];
      else if (m < 54) v = scrw[(m - 36) * 512 + k];
      wc[j] = (_Float16)v;
    } else if (i < R0 + R1 + R2 + R3) {
      int m = (int)(i - R0 - R1 - R2);
      float v = 0.f;
      if (m < 36)      v = locb[m];
      else if (m < 54) v = scrb[m - 36];
      bc[m] = v;
    } else if (i < R0 + R1 + R2 + R3 + R4) {
      int c = (int)(i - R0 - R1 - R2 - R3);
      csc[c] = -__builtin_inff();
      cbox[c * 4 + 0] = 0.f; cbox[c * 4 + 1] = 0.f;
      cbox[c * 4 + 2] = 0.f; cbox[c * 4 + 3] = 0.f;
    } else if (i < R0 + R1 + R2 + R3 + R4 + R5) {
      hist[(int)(i - R0 - R1 - R2 - R3 - R4)] = 0;
    } else {
      misc[(int)(i - R0 - R1 - R2 - R3 - R4 - R5)] = 0;
    }
  }
}

// ---------------------------------------------------------------------------
// Kernel 1: 3x3 conv 512->512 + bias + ReLU as implicit GEMM with WMMA f16.
// WG = 256 thr (8 waves); tile = 128 co x 128 px (one image row); K = 9*512.
// Double-buffered LDS, tiles staged with async global->LDS DMA that overlaps
// the previous step's WMMAs (counted s_wait_asynccnt; in-order completion).
// ---------------------------------------------------------------------------
__global__ void rpn_conv3x3_wmma(const _Float16* __restrict__ in_t,
                                 const _Float16* __restrict__ w_t,
                                 const float* __restrict__ conv_b,
                                 const _Float16* __restrict__ zpad,
                                 _Float16* __restrict__ H_t) {
  __shared__ __attribute__((aligned(32))) _Float16 Alds[2][128 * 32]; // [co][k]
  __shared__ __attribute__((aligned(32))) _Float16 Blds[2][128 * 32]; // [px][k]

  const int t = threadIdx.x;
  const int wave = t >> 5, lane = t & 31;
  const int wm = wave >> 1, wn = wave & 1;   // wave tile: 32 co x 64 px
  const int h = lane >> 4, lm = lane & 15;
  const int y = blockIdx.x;                  // pixel row; n0 = y*128
  const int co0 = blockIdx.y * 128;
  constexpr int kSteps = 9 * 16;             // r-major, 16 k-steps of 32 per tap

  // stage one (r,k0) tile pair into buffer `buf`; 4 async ops per wave, always
  // uniform (halo lanes read a 16B zero pad instead of diverging).
  auto issue = [&](int buf, int s) {
    const int r = s >> 4, k0 = (s & 15) << 5;
    const int dy = r / 3 - 1, dx = r % 3 - 1;
    const int yy = y + dy;
    const bool yok = (unsigned)yy < (unsigned)kFH;
    _Float16* Al = Alds[buf];
    _Float16* Bl = Blds[buf];
#pragma unroll
    for (int c = 0; c < 2; ++c) {            // A: 128 co rows x 64B weights
      const int idx = t + c * 256;
      const int row = idx >> 2, ch = idx & 3;
      async_load_b128(lds_lo32(Al + row * 32 + ch * 8),
                      w_t + (size_t)(co0 + row) * kK9 + r * 512 + k0 + ch * 8);
    }
#pragma unroll
    for (int c = 0; c < 2; ++c) {            // B: 128 px x 64B shifted activations
      const int idx = t + c * 256;
      const int px = idx >> 2, ch = idx & 3;
      const int xx = px + dx;
      const bool ok = yok && (unsigned)xx < (unsigned)kFW;
      const _Float16* src =
          ok ? (in_t + ((size_t)yy * kFW + xx) * 512 + k0 + ch * 8) : zpad;
      async_load_b128(lds_lo32(Bl + px * 32 + ch * 8), src);
    }
  };

  v8f acc[2][4];
#pragma unroll
  for (int a = 0; a < 2; ++a)
#pragma unroll
    for (int b = 0; b < 4; ++b) acc[a][b] = (v8f){0.f,0.f,0.f,0.f,0.f,0.f,0.f,0.f};

  issue(0, 0);
  for (int s = 0; s < kSteps; ++s) {
    const int cur = s & 1;
    if (s + 1 < kSteps) {
      issue(cur ^ 1, s + 1);   // next tile DMA overlaps this step's WMMAs
      WAIT_ASYNC(4);           // in-order: <=4 outstanding => step-s tiles done
    } else {
      WAIT_ASYNC(0);
    }
    __syncthreads();

    // A fragments: lane lm = row M, half-wave h selects K range [h*16, h*16+16)
    v16h af[2];
#pragma unroll
    for (int tm = 0; tm < 2; ++tm) {
      const _Float16* p = Alds[cur] + (wm * 32 + tm * 16 + lm) * 32;
      v8h lo = *(const v8h*)(p + h * 16);
      v8h hi = *(const v8h*)(p + h * 16 + 8);
#pragma unroll
      for (int i = 0; i < 8; ++i) { af[tm][i] = lo[i]; af[tm][i + 8] = hi[i]; }
    }
    // B fragments: lane lm = col N, half-wave h selects K range [h*16, h*16+16)
    v16h bf[4];
#pragma unroll
    for (int tn = 0; tn < 4; ++tn)
      bf[tn] = *(const v16h*)(Blds[cur] + (wn * 64 + tn * 16 + lm) * 32 + h * 16);

#pragma unroll
    for (int tm = 0; tm < 2; ++tm)
#pragma unroll
      for (int tn = 0; tn < 4; ++tn)
        acc[tm][tn] = __builtin_amdgcn_wmma_f32_16x16x32_f16(
            false, af[tm], false, bf[tn], (short)0, acc[tm][tn], false, false);

    __syncthreads();           // all reads of buf `cur` done before s+2 overwrites
  }

  // epilogue: bias + ReLU, pack 8 consecutive channels to f16, store [pix][co]
#pragma unroll
  for (int tm = 0; tm < 2; ++tm) {
    const int cob = co0 + wm * 32 + tm * 16 + 8 * h;   // D: m = j + 8h
    const float4 b0 = *(const float4*)(conv_b + cob);
    const float4 b1 = *(const float4*)(conv_b + cob + 4);
    const float bias[8] = {b0.x, b0.y, b0.z, b0.w, b1.x, b1.y, b1.z, b1.w};
#pragma unroll
    for (int tn = 0; tn < 4; ++tn) {
      const int n = y * kFW + wn * 64 + tn * 16 + lm;  // D: n = lane&15
      v8h outv;
#pragma unroll
      for (int j = 0; j < 8; ++j) {
        float v = acc[tm][tn][j] + bias[j];
        outv[j] = (_Float16)(v > 0.f ? v : 0.f);
      }
      *(v8h*)(H_t + (size_t)n * 512 + cob) = outv;
    }
  }
}

// ---------------------------------------------------------------------------
// Kernel 2: fused 1x1 loc(36)+score(18) convs, M padded to 64, WMMA f16.
// Same async double-buffered staging (3 async ops per wave per step).
// ---------------------------------------------------------------------------
__global__ void rpn_head_wmma(const _Float16* __restrict__ H_t,
                              const _Float16* __restrict__ wc,
                              const float* __restrict__ bc,
                              float* __restrict__ loc_out,
                              float* __restrict__ scr_out) {
  __shared__ __attribute__((aligned(32))) _Float16 Alds[2][64 * 32];
  __shared__ __attribute__((aligned(32))) _Float16 Blds[2][128 * 32];

  const int t = threadIdx.x;
  const int wave = t >> 5, lane = t & 31;
  const int wm = wave >> 1, wn = wave & 1;
  const int h = lane >> 4, lm = lane & 15;
  const int n0 = blockIdx.x * 128;
  constexpr int kSteps = 512 / 32;

  auto issue = [&](int buf, int s) {
    const int k0 = s << 5;
    {
      const int row = t >> 2, ch = t & 3;    // 64 rows x 4 chunks: 1 per thread
      async_load_b128(lds_lo32(Alds[buf] + row * 32 + ch * 8),
                      wc + (size_t)row * 512 + k0 + ch * 8);
    }
#pragma unroll
    for (int c = 0; c < 2; ++c) {
      const int idx = t + c * 256;
      const int px = idx >> 2, ch = idx & 3;
      async_load_b128(lds_lo32(Blds[buf] + px * 32 + ch * 8),
                      H_t + (size_t)(n0 + px) * 512 + k0 + ch * 8);
    }
  };

  v8f acc[4];
#pragma unroll
  for (int b = 0; b < 4; ++b) acc[b] = (v8f){0.f,0.f,0.f,0.f,0.f,0.f,0.f,0.f};

  issue(0, 0);
  for (int s = 0; s < kSteps; ++s) {
    const int cur = s & 1;
    if (s + 1 < kSteps) {
      issue(cur ^ 1, s + 1);
      WAIT_ASYNC(3);
    } else {
      WAIT_ASYNC(0);
    }
    __syncthreads();

    v16h af;
    {
      const _Float16* p = Alds[cur] + (wm * 16 + lm) * 32;
      v8h lo = *(const v8h*)(p + h * 16);
      v8h hi = *(const v8h*)(p + h * 16 + 8);
#pragma unroll
      for (int i = 0; i < 8; ++i) { af[i] = lo[i]; af[i + 8] = hi[i]; }
    }
    v16h bf[4];
#pragma unroll
    for (int tn = 0; tn < 4; ++tn)
      bf[tn] = *(const v16h*)(Blds[cur] + (wn * 64 + tn * 16 + lm) * 32 + h * 16);
#pragma unroll
    for (int tn = 0; tn < 4; ++tn)
      acc[tn] = __builtin_amdgcn_wmma_f32_16x16x32_f16(
          false, af, false, bf[tn], (short)0, acc[tn], false, false);

    __syncthreads();
  }

#pragma unroll
  for (int tn = 0; tn < 4; ++tn) {
    const int n = n0 + wn * 64 + tn * 16 + lm;
#pragma unroll
    for (int j = 0; j < 8; ++j) {
      const int m = wm * 16 + 8 * h + j;
      const float v = acc[tn][j] + bc[m];
      if (m < 36)      loc_out[(size_t)n * 36 + m] = v;
      else if (m < 54) scr_out[(size_t)n * 18 + (m - 36)] = v;
    }
  }
}

// ---------------------------------------------------------------------------
// Kernel 3: per-anchor: fg score, decode+clip, min-size filter, score hist,
// and both losses (block-reduced, 3 atomics/block).
// ---------------------------------------------------------------------------
__global__ void rpn_anchor(const float* __restrict__ loc_out,
                           const float* __restrict__ scr_out,
                           const float* __restrict__ anch,
                           const float* __restrict__ gtloc,
                           const int* __restrict__ gtlab,
                           float* __restrict__ boxes, float* __restrict__ sc,
                           int* __restrict__ hist, float* __restrict__ mf) {
  const int a = blockIdx.x * 256 + threadIdx.x;     // grid sized exactly
  const int t = threadIdx.x;
  float sl1 = 0.f, nll = 0.f, nv = 0.f;

  const int n = a / 9, j = a - n * 9;
  const float* lp = loc_out + (size_t)n * 36 + j * 4;
  const float l0 = lp[0], l1 = lp[1], l2 = lp[2], l3 = lp[3];
  const float s0 = scr_out[(size_t)n * 18 + j * 2];
  const float s1 = scr_out[(size_t)n * 18 + j * 2 + 1];
  const float fg = 1.f / (1.f + __expf(s0 - s1));

  const float* an = anch + (size_t)a * 4;
  const float ha = an[2] - an[0], wa = an[3] - an[1];
  const float cya = an[0] + 0.5f * ha, cxa = an[1] + 0.5f * wa;
  const float cy = l0 * ha + cya, cx = l1 * wa + cxa;
  const float hh = __expf(l2) * ha, ww = __expf(l3) * wa;
  const float b0 = fminf(fmaxf(cy - 0.5f * hh, 0.f), 2048.f);
  const float b1 = fminf(fmaxf(cx - 0.5f * ww, 0.f), 2048.f);
  const float b2 = fminf(fmaxf(cy + 0.5f * hh, 0.f), 2048.f);
  const float b3 = fminf(fmaxf(cx + 0.5f * ww, 0.f), 2048.f);
  boxes[(size_t)a * 4 + 0] = b0; boxes[(size_t)a * 4 + 1] = b1;
  boxes[(size_t)a * 4 + 2] = b2; boxes[(size_t)a * 4 + 3] = b3;
  const bool keep = ((b2 - b0) >= 16.f) && ((b3 - b1) >= 16.f);
  sc[a] = keep ? fg : -__builtin_inff();
  if (keep) {
    int bin = (int)(fg * 1024.f);
    bin = bin < 0 ? 0 : (bin > 1023 ? 1023 : bin);
    atomicAdd(&hist[bin], 1);
  }

  const int lbl = gtlab[a];
  if (lbl >= 0) {
    nv = 1.f;
    const float mx = fmaxf(s0, s1);
    const float lse = mx + __logf(__expf(s0 - mx) + __expf(s1 - mx));
    nll = lse - (lbl > 0 ? s1 : s0);
    if (lbl > 0) {
      const float* g = gtloc + (size_t)a * 4;
      const float lv[4] = {l0, l1, l2, l3};
#pragma unroll
      for (int c = 0; c < 4; ++c) {
        const float d = lv[c] - g[c];
        const float ad = fabsf(d);
        sl1 += (ad < (1.f / 9.f)) ? 4.5f * d * d : (ad - 0.5f / 9.f);
      }
    }
  }

  __shared__ float rA[256], rB[256], rC[256];
  rA[t] = sl1; rB[t] = nll; rC[t] = nv;
  __syncthreads();
  for (int off = 128; off > 0; off >>= 1) {
    if (t < off) { rA[t] += rA[t + off]; rB[t] += rB[t + off]; rC[t] += rC[t + off]; }
    __syncthreads();
  }
  if (t == 0) {
    atomicAdd(&mf[2], rA[0]);
    atomicAdd(&mf[3], rB[0]);
    atomicAdd(&mf[4], rC[0]);
  }
}

// ---------------------------------------------------------------------------
// Kernel 4: histogram suffix scan -> score-bin threshold for ~top-12000
// ---------------------------------------------------------------------------
__global__ void rpn_thresh(const int* __restrict__ hist, int* __restrict__ misc) {
  int cum = 0, thr = 0;
  for (int b = 1023; b >= 0; --b) {
    cum += hist[b];
    if (cum >= kPre) { thr = b; break; }
  }
  misc[1] = thr;
}

// ---------------------------------------------------------------------------
// Kernel 5: compact candidates above threshold (capped at 12000)
// ---------------------------------------------------------------------------
__global__ void rpn_compact(const float* __restrict__ boxes, const float* __restrict__ sc,
                            int* __restrict__ misc, float* __restrict__ cbox,
                            float* __restrict__ csc) {
  const int a = blockIdx.x * 256 + threadIdx.x;
  const float s = sc[a];
  if (s > -1e30f) {
    int bin = (int)(s * 1024.f);
    bin = bin < 0 ? 0 : (bin > 1023 ? 1023 : bin);
    if (bin >= misc[1]) {
      const int pos = atomicAdd(&misc[0], 1);
      if (pos < kPre) {
        cbox[(size_t)pos * 4 + 0] = boxes[(size_t)a * 4 + 0];
        cbox[(size_t)pos * 4 + 1] = boxes[(size_t)a * 4 + 1];
        cbox[(size_t)pos * 4 + 2] = boxes[(size_t)a * 4 + 2];
        cbox[(size_t)pos * 4 + 3] = boxes[(size_t)a * 4 + 3];
        csc[pos] = s;
      }
    }
  }
}

// ---------------------------------------------------------------------------
// Kernel 6: greedy NMS, single WGP; scores+boxes LDS-resident (~248KB of the
// 320KB CDNA5 WGP LDS). 2000 iterative argmax + suppress.
// ---------------------------------------------------------------------------
__global__ void rpn_nms(const float* __restrict__ cbox, const float* __restrict__ csc,
                        float* __restrict__ out) {
  extern __shared__ float sm[];
  float*  s   = sm;                               // 12000 (pad to 12032)
  float4* bx  = (float4*)(sm + 12032);            // 12000 float4 = 48000 floats
  float*  rv  = sm + 12032 + 48000;               // 1024
  int*    ri  = (int*)(sm + 12032 + 48000 + 1024);// 1024
  const int t = threadIdx.x;

  for (int e = t; e < kPre; e += 1024) {
    s[e] = csc[e];
    bx[e] = *(const float4*)(cbox + (size_t)e * 4);
  }
  __syncthreads();

  for (int i = 0; i < kPost; ++i) {
    float bv = -__builtin_inff(); int bi = 0;
    for (int e = t; e < kPre; e += 1024)
      if (s[e] > bv) { bv = s[e]; bi = e; }
    rv[t] = bv; ri[t] = bi;
    __syncthreads();
    for (int off = 512; off > 0; off >>= 1) {
      if (t < off && rv[t + off] > rv[t]) { rv[t] = rv[t + off]; ri[t] = ri[t + off]; }
      __syncthreads();
    }
    const int j = ri[0];
    const float jv = rv[0];
    const bool ok = jv > -1e30f;
    const float4 jb = bx[j];
    if (t == 0)
      *(float4*)(out + (size_t)i * 4) = ok ? jb : make_float4(0.f, 0.f, 0.f, 0.f);
    const float ja = (jb.z - jb.x) * (jb.w - jb.y);
    for (int e = t; e < kPre; e += 1024) {
      if (!ok) { s[e] = -__builtin_inff(); continue; }
      const float4 b = bx[e];
      const float yy1 = fmaxf(jb.x, b.x), xx1 = fmaxf(jb.y, b.y);
      const float yy2 = fminf(jb.z, b.z), xx2 = fminf(jb.w, b.w);
      const float inter = fmaxf(yy2 - yy1, 0.f) * fmaxf(xx2 - xx1, 0.f);
      const float area = (b.z - b.x) * (b.w - b.y);
      const float iou = inter / (ja + area - inter + 1e-9f);
      if (iou > 0.7f || e == j) s[e] = -__builtin_inff();
    }
    __syncthreads();
  }
}

// ---------------------------------------------------------------------------
// Kernel 7: finalize losses
// ---------------------------------------------------------------------------
__global__ void rpn_finalize(const float* __restrict__ mf, float* __restrict__ out) {
  const float nv = mf[4];
  out[8000] = mf[2] / nv;
  out[8001] = mf[3] / nv;
}

// ---------------------------------------------------------------------------
extern "C" void kernel_launch(void* const* d_in, const int* in_sizes, int n_in,
                              void* d_out, int out_size, void* d_ws, size_t ws_size,
                              hipStream_t stream) {
  (void)in_sizes; (void)n_in; (void)out_size; (void)ws_size;
  const float* feat  = (const float*)d_in[0];
  const float* convw = (const float*)d_in[1];
  const float* convb = (const float*)d_in[2];
  const float* locw  = (const float*)d_in[3];
  const float* locb  = (const float*)d_in[4];
  const float* scrw  = (const float*)d_in[5];
  const float* scrb  = (const float*)d_in[6];
  const float* gtloc = (const float*)d_in[7];
  const int*   gtlab = (const int*)d_in[8];
  const float* anch  = (const float*)d_in[9];

  char* ws = (char*)d_ws;
  _Float16* in_t    = (_Float16*)(ws + OFF_IN_T);
  _Float16* w_t     = (_Float16*)(ws + OFF_W_T);
  _Float16* wc      = (_Float16*)(ws + OFF_WC);
  float*    bc      = (float*)(ws + OFF_BC);
  _Float16* H_t     = (_Float16*)(ws + OFF_H_T);
  float*    loc_out = (float*)(ws + OFF_LOC);
  float*    scr_out = (float*)(ws + OFF_SCR);
  float*    boxes   = (float*)(ws + OFF_BOX);
  float*    scv     = (float*)(ws + OFF_SC);
  float*    cbox    = (float*)(ws + OFF_CBOX);
  float*    csc     = (float*)(ws + OFF_CSC);
  int*      hist    = (int*)(ws + OFF_HIST);
  int*      mi      = (int*)(ws + OFF_MISC);
  float*    mf      = (float*)(ws + OFF_MISC);
  const _Float16* zpad = (const _Float16*)(ws + OFF_MISC + 32);  // 16B of zeros
  float*    out     = (float*)d_out;

  rpn_prep<<<8192, 256, 0, stream>>>(feat, convw, locw, scrw, locb, scrb,
                                     in_t, w_t, wc, bc, cbox, csc, hist, mi);
  rpn_conv3x3_wmma<<<dim3(128, 4), 256, 0, stream>>>(in_t, w_t, convb, zpad, H_t);
  rpn_head_wmma<<<128, 256, 0, stream>>>(H_t, wc, bc, loc_out, scr_out);
  rpn_anchor<<<kNAnch / 256, 256, 0, stream>>>(loc_out, scr_out, anch, gtloc, gtlab,
                                               boxes, scv, hist, mf);
  rpn_thresh<<<1, 1, 0, stream>>>(hist, mi);
  rpn_compact<<<kNAnch / 256, 256, 0, stream>>>(boxes, scv, mi, cbox, csc);
  const size_t nms_smem = (size_t)(12032 + 48000 + 1024 + 1024 + 4) * sizeof(float);
  rpn_nms<<<1, 1024, nms_smem, stream>>>(cbox, csc, out);
  rpn_finalize<<<1, 1, 0, stream>>>(mf, out);
}
